// SparseAttention3dv2_2972117369407
// MI455X (gfx1250) — matrix operand order repaired and checked by reading, not compiled, not run on hardware
//
#include <hip/hip_runtime.h>
#include <hip/hip_bf16.h>

// Problem constants (from reference)
#define NQ   20000
#define MV   80000
#define KK   48
#define CC   64
#define FFd  256
#define HH   4
#define HD   16
#define OUTC 64

typedef unsigned short u16;
typedef __attribute__((ext_vector_type(16))) __bf16 v16bf;
typedef __attribute__((ext_vector_type(8)))  float  v8f;

// ---- bf16 helpers -----------------------------------------------------------
__device__ __forceinline__ u16 f2bf(float f) {
    unsigned u = __builtin_bit_cast(unsigned, f);
    u += 0x7FFFu + ((u >> 16) & 1u);     // round-to-nearest-even
    return (u16)(u >> 16);
}

union V16U { uint4 u[2]; v16bf v; };

// A operand (16x32 bf16) from LDS, row-major [16][ldk] bf16 tile.
// ISA layout: lanes 0-15 hold row m, K = kc+[0..7] and kc+[16..23];
//             lanes 16-31 hold row m, K = kc+[8..15] and kc+[24..31].
__device__ __forceinline__ v16bf load_A_lds(const u16* base, int ldk, int lane, int kc) {
    int row = lane & 15, half = lane >> 4;
    const u16* r = base + row * ldk + kc + half * 8;
    V16U t;
    t.u[0] = *(const uint4*)(r);
    t.u[1] = *(const uint4*)(r + 16);
    return t.v;
}

// B operand (32x16 bf16) from global bf16 weights, row-major [ncols][ld]:
// B(k,n) = W[n][k]; lane half h reads K = kc+8h+[0..7] and kc+16+8h+[0..7].
__device__ __forceinline__ v16bf load_B_g(const u16* w, int ld, int lane, int ntile, int kc) {
    const u16* r = w + (size_t)((ntile << 4) + (lane & 15)) * ld + kc + ((lane >> 4) << 3);
    V16U t;
    t.u[0] = *(const uint4*)(r);
    t.u[1] = *(const uint4*)(r + 16);
    return t.v;
}

__device__ __forceinline__ v8f wmma_bf(v16bf a, v16bf b, v8f c) {
    // D = A(16x32 bf16) x B(32x16 bf16) + C(16x16 f32)
    return __builtin_amdgcn_wmma_f32_16x16x32_bf16(false, a, false, b, (short)0, c, false, false);
}

// Single 16x16 output tile, split-K across NACC independent accumulator chains
// (breaks the WMMA->WMMA RAW chain; bf16 WMMA wants ~4 independent ops between).
template <int KDIM, int NACC>
__device__ __forceinline__ v8f gemm_one(const u16* aLds, int aLd, const u16* wG, int wLd,
                                        int ntile, int lane, float biasv) {
    v8f acc[NACC];
#pragma unroll
    for (int j = 0; j < NACC; ++j)
#pragma unroll
        for (int i = 0; i < 8; ++i) acc[j][i] = (j == 0) ? biasv : 0.f;
    int which = 0;
#pragma unroll
    for (int kc = 0; kc < KDIM; kc += 32) {
        v16bf a = load_A_lds(aLds, aLd, lane, kc);
        v16bf b = load_B_g(wG, wLd, lane, ntile, kc);
        acc[which] = wmma_bf(a, b, acc[which]);
        which = (which + 1) % NACC;
    }
#pragma unroll
    for (int j = 1; j < NACC; ++j)
#pragma unroll
        for (int i = 0; i < 8; ++i) acc[0][i] += acc[j][i];
    return acc[0];
}

// Four independent 16x16 output tiles sharing the A operand: one A load per
// k-chunk feeds 4 WMMAs, giving dependency distance 4 between same-acc WMMAs.
__device__ __forceinline__ void gemm_quad(
    const u16* aLds, int aLd, int wLd, int kdim, int lane,
    const u16* wA, int ntA, float bA, const u16* wB, int ntB, float bB,
    const u16* wC, int ntC, float bC, const u16* wD, int ntD, float bD,
    v8f& oA, v8f& oB, v8f& oC, v8f& oD) {
#pragma unroll
    for (int i = 0; i < 8; ++i) { oA[i] = bA; oB[i] = bB; oC[i] = bC; oD[i] = bD; }
    for (int kc = 0; kc < kdim; kc += 32) {
        v16bf a = load_A_lds(aLds, aLd, lane, kc);
        v16bf b0 = load_B_g(wA, wLd, lane, ntA, kc);
        v16bf b1 = load_B_g(wB, wLd, lane, ntB, kc);
        v16bf b2 = load_B_g(wC, wLd, lane, ntC, kc);
        v16bf b3 = load_B_g(wD, wLd, lane, ntD, kc);
        oA = wmma_bf(a, b0, oA);
        oB = wmma_bf(a, b1, oB);
        oC = wmma_bf(a, b2, oC);
        oD = wmma_bf(a, b3, oD);
    }
}

// D layout: VGPR r, lanes 0-15 -> (row r,  col lane), lanes 16-31 -> (row r+8, col lane-16)
__device__ __forceinline__ void store_D_f32(float* lds, int ldc, int lane, int ntile, v8f d) {
    int col = (ntile << 4) + (lane & 15);
    int roff = (lane >> 4) * 8;
#pragma unroll
    for (int r = 0; r < 8; ++r) lds[(r + roff) * ldc + col] = d[r];
}

__device__ __forceinline__ void store_D_relu_bf16(u16* lds, int ldc, int lane, int ntile, v8f d) {
    int col = (ntile << 4) + (lane & 15);
    int roff = (lane >> 4) * 8;
#pragma unroll
    for (int r = 0; r < 8; ++r) {
        float v = d[r];
        v = v > 0.f ? v : 0.f;
        lds[(r + roff) * ldc + col] = f2bf(v);
    }
}

// ---- prep: f32 -> bf16 weight conversion -----------------------------------
// wb layout (u16 offsets): Wq 0, Wk 4096, Wv 8192, Wo 12288, outl 16384,
//                          lin1 20480 (16384), lin2 36864 (16384)
__global__ __launch_bounds__(256) void prep_kernel(
    const float* __restrict__ in_w, const float* __restrict__ out_w,
    const float* __restrict__ lin1_w, const float* __restrict__ lin2_w,
    const float* __restrict__ outl_w, u16* __restrict__ wb) {
    int t = blockIdx.x * 256 + threadIdx.x;
    int stride = gridDim.x * 256;
    for (int i = t; i < 3 * CC * CC; i += stride) wb[i] = f2bf(in_w[i]);           // Wq|Wk|Wv
    for (int i = t; i < CC * CC; i += stride) wb[12288 + i] = f2bf(out_w[i]);
    for (int i = t; i < OUTC * CC; i += stride) wb[16384 + i] = f2bf(outl_w[i]);
    for (int i = t; i < FFd * CC; i += stride) wb[20480 + i] = f2bf(lin1_w[i]);
    for (int i = t; i < CC * FFd; i += stride) wb[36864 + i] = f2bf(lin2_w[i]);
}

// ---- fused attention + FFN over a 16-query tile ----------------------------
__global__ __launch_bounds__(128) void attn_fused_kernel(
    const float* __restrict__ vf, const float* __restrict__ vc,
    const float* __restrict__ qc, const int* __restrict__ kidx,
    const float* __restrict__ q_w, const float* __restrict__ q_b,
    const float* __restrict__ kpw, const float* __restrict__ kpb,
    const float* __restrict__ in_b, const float* __restrict__ out_b,
    const float* __restrict__ lin1_b, const float* __restrict__ lin2_b,
    const u16* __restrict__ WqB, const u16* __restrict__ WkB,
    const u16* __restrict__ WvB, const u16* __restrict__ WoB,
    const u16* __restrict__ l1B, const u16* __restrict__ l2B,
    float* __restrict__ X) {
    __shared__ __align__(16) u16 qf[16][CC];        // bf16 staging (q_feat / ctx / att)
    __shared__ float qS[16][CC];                    // scaled q
    __shared__ __align__(16) u16 pS[4][16][CC];     // per-wave key_feats tile (bf16)
    __shared__ float kkS[4][16][CC];
    __shared__ float vvS[4][16][CC];
    __shared__ int idxS[4][16];
    __shared__ float ctxS[16][CC];                  // ctx, later lin2 output
    __shared__ float attF[16][CC];
    __shared__ __align__(16) u16 hidB[16][FFd];     // relu(lin1) bf16
    __shared__ __align__(16) float4 kpwS[CC];       // (kpos_w row, kpos_b) packed

    const int tid = threadIdx.x;
    const int lane = tid & 31, wave = tid >> 5;
    const int n0 = blockIdx.x * 16;

    // --- stage kpos weights, compute q_feat = relu(qc @ q_w.T + q_b) ---
    if (tid < CC) {
        kpwS[tid] = make_float4(kpw[tid * 3 + 0], kpw[tid * 3 + 1], kpw[tid * 3 + 2], kpb[tid]);
    }
    {
        int row = tid >> 3, c0 = (tid & 7) * 8;
        int n = n0 + row;
        float x = qc[n * 3 + 0], y = qc[n * 3 + 1], z = qc[n * 3 + 2];
#pragma unroll
        for (int j = 0; j < 8; ++j) {
            int c = c0 + j;
            float v = q_w[c * 3 + 0] * x + q_w[c * 3 + 1] * y + q_w[c * 3 + 2] * z + q_b[c];
            qf[row][c] = f2bf(v > 0.f ? v : 0.f);
        }
    }
    __syncthreads();
    // --- q = (q_feat @ Wq.T + bq) / sqrt(hd) ---
    {
        int nt = wave;
        float bq = in_b[(nt << 4) + (lane & 15)];
        v8f a = gemm_one<CC, 2>(&qf[0][0], CC, WqB, CC, nt, lane, bq);
#pragma unroll
        for (int i = 0; i < 8; ++i) a[i] *= 0.25f;   // 1/sqrt(16)
        store_D_f32(&qS[0][0], CC, lane, nt, a);
    }
    __syncthreads();

    // --- per-lane loop-invariants for the gather phase ---
    const int grow = lane & 15, ghalf = lane >> 4;
    const float gqx = qc[(n0 + grow) * 3 + 0];
    const float gqy = qc[(n0 + grow) * 3 + 1];
    const float gqz = qc[(n0 + grow) * 3 + 2];

    // --- online-softmax state (threads 0..63: one (query,head) each) ---
    const int qrow = tid & 15, head = (tid >> 4) & 3;
    float m = -1e30f, lsum = 0.f;
    float ctx[HD], qreg[HD];
#pragma unroll
    for (int d = 0; d < HD; ++d) ctx[d] = 0.f;
    if (tid < 64) {
#pragma unroll
        for (int d = 0; d < HD; ++d) qreg[d] = qS[qrow][head * HD + d];
    }

    for (int s0 = 0; s0 < KK; s0 += 4) {
        int slot = s0 + wave;
        // gather + positional MLP -> key_feats tile, bf16 (wave-private)
        {
            int id = kidx[(size_t)(n0 + grow) * KK + slot];
            if (ghalf == 0) idxS[wave][grow] = id;
            int safe = id < 0 ? 0 : id;
            float rx = vc[safe * 3 + 0] - gqx;
            float ry = vc[safe * 3 + 1] - gqy;
            float rz = vc[safe * 3 + 2] - gqz;
            const float* vrow = vf + (size_t)safe * CC;
#pragma unroll
            for (int g = 0; g < 8; ++g) {
                int c = ghalf * 32 + g * 4;
                float4 f = *(const float4*)(vrow + c);
                float4 w0 = kpwS[c + 0], w1 = kpwS[c + 1], w2 = kpwS[c + 2], w3 = kpwS[c + 3];
                float p0 = fmaxf(w0.x * rx + w0.y * ry + w0.z * rz + w0.w, 0.f);
                float p1 = fmaxf(w1.x * rx + w1.y * ry + w1.z * rz + w1.w, 0.f);
                float p2 = fmaxf(w2.x * rx + w2.y * ry + w2.z * rz + w2.w, 0.f);
                float p3 = fmaxf(w3.x * rx + w3.y * ry + w3.z * rz + w3.w, 0.f);
                ushort4 o;
                o.x = f2bf(f.x + p0); o.y = f2bf(f.y + p1);
                o.z = f2bf(f.z + p2); o.w = f2bf(f.w + p3);
                *(ushort4*)&pS[wave][grow][c] = o;
            }
        }
        __syncthreads();
        // K/V projections for this wave's slot: quad WMMA chains sharing A loads
        for (int ntp = 0; ntp < 2; ++ntp) {
            int nt0 = ntp * 2, nt1 = ntp * 2 + 1;
            float bk0 = in_b[CC + (nt0 << 4) + (lane & 15)];
            float bv0 = in_b[2 * CC + (nt0 << 4) + (lane & 15)];
            float bk1 = in_b[CC + (nt1 << 4) + (lane & 15)];
            float bv1 = in_b[2 * CC + (nt1 << 4) + (lane & 15)];
            v8f k0, v0, k1, v1;
            gemm_quad(&pS[wave][0][0], CC, CC, CC, lane,
                      WkB, nt0, bk0, WvB, nt0, bv0,
                      WkB, nt1, bk1, WvB, nt1, bv1, k0, v0, k1, v1);
            store_D_f32(&kkS[wave][0][0], CC, lane, nt0, k0);
            store_D_f32(&vvS[wave][0][0], CC, lane, nt0, v0);
            store_D_f32(&kkS[wave][0][0], CC, lane, nt1, k1);
            store_D_f32(&vvS[wave][0][0], CC, lane, nt1, v1);
        }
        __syncthreads();
        // flash-style update over the 4 slots just produced
        if (tid < 64) {
            for (int s = 0; s < 4; ++s) {
                if (idxS[s][qrow] < 0) continue;     // masked (padded) key
                const float* kr = &kkS[s][qrow][head * HD];
                const float* vr = &vvS[s][qrow][head * HD];
                float sc = 0.f;
#pragma unroll
                for (int d = 0; d < HD; ++d) sc += qreg[d] * kr[d];
                float nm = sc > m ? sc : m;
                float e0 = __expf(m - nm), e1 = __expf(sc - nm);
                lsum = lsum * e0 + e1;
#pragma unroll
                for (int d = 0; d < HD; ++d) ctx[d] = ctx[d] * e0 + e1 * vr[d];
                m = nm;
            }
        }
        __syncthreads();
    }
    if (tid < 64) {
        float inv = 1.f / lsum;
#pragma unroll
        for (int d = 0; d < HD; ++d) ctxS[qrow][head * HD + d] = ctx[d] * inv;
    }
    __syncthreads();

    // --- att = ctx @ out_w.T + out_b ---
    {
        int row = tid >> 3, c0 = (tid & 7) * 8;
#pragma unroll
        for (int j = 0; j < 8; ++j) qf[row][c0 + j] = f2bf(ctxS[row][c0 + j]);
    }
    __syncthreads();
    {
        int nt = wave;
        float bo = out_b[(nt << 4) + (lane & 15)];
        v8f a = gemm_one<CC, 2>(&qf[0][0], CC, WoB, CC, nt, lane, bo);
        store_D_f32(&attF[0][0], CC, lane, nt, a);
    }
    __syncthreads();
    {
        int row = tid >> 3, c0 = (tid & 7) * 8;
#pragma unroll
        for (int j = 0; j < 8; ++j) qf[row][c0 + j] = f2bf(attF[row][c0 + j]);
    }
    __syncthreads();
    // --- hidden = relu(att @ lin1_w.T + lin1_b): 4 independent n-tiles/wave ---
    {
        int nt = wave * 4;
        float b0 = lin1_b[((nt + 0) << 4) + (lane & 15)];
        float b1 = lin1_b[((nt + 1) << 4) + (lane & 15)];
        float b2 = lin1_b[((nt + 2) << 4) + (lane & 15)];
        float b3 = lin1_b[((nt + 3) << 4) + (lane & 15)];
        v8f h0, h1, h2, h3;
        gemm_quad(&qf[0][0], CC, CC, CC, lane,
                  l1B, nt + 0, b0, l1B, nt + 1, b1,
                  l1B, nt + 2, b2, l1B, nt + 3, b3, h0, h1, h2, h3);
        store_D_relu_bf16(&hidB[0][0], FFd, lane, nt + 0, h0);
        store_D_relu_bf16(&hidB[0][0], FFd, lane, nt + 1, h1);
        store_D_relu_bf16(&hidB[0][0], FFd, lane, nt + 2, h2);
        store_D_relu_bf16(&hidB[0][0], FFd, lane, nt + 3, h3);
    }
    __syncthreads();
    // --- lin2: hidden(16x256) @ lin2_w.T, 1 n-tile per wave, 4-way split-K ---
    {
        int nt = wave;
        float b2 = lin2_b[(nt << 4) + (lane & 15)];
        v8f a = gemm_one<FFd, 4>(&hidB[0][0], FFd, l2B, FFd, nt, lane, b2);
        store_D_f32(&ctxS[0][0], CC, lane, nt, a);
    }
    __syncthreads();
    // --- x = att + ffn -> workspace X ---
    {
        int row = tid >> 3, c0 = (tid & 7) * 8;
#pragma unroll
        for (int j = 0; j < 8; ++j) {
            int c = c0 + j;
            X[(size_t)(n0 + row) * CC + c] = attF[row][c] + ctxS[row][c];
        }
    }
}

// ---- deterministic per-channel reduction (sum, sum of squares) -------------
__global__ __launch_bounds__(256) void colsum_kernel(const float* __restrict__ A,
                                                     float* __restrict__ sumOut,
                                                     float* __restrict__ ssqOut) {
    int c = blockIdx.x;
    float s = 0.f, q = 0.f;
    for (int r = threadIdx.x; r < NQ; r += 256) {
        float v = A[(size_t)r * CC + c];
        s += v; q += v * v;
    }
    __shared__ float sh[256], sh2[256];
    sh[threadIdx.x] = s; sh2[threadIdx.x] = q;
    __syncthreads();
    for (int off = 128; off > 0; off >>= 1) {
        if (threadIdx.x < off) {
            sh[threadIdx.x] += sh[threadIdx.x + off];
            sh2[threadIdx.x] += sh2[threadIdx.x + off];
        }
        __syncthreads();
    }
    if (threadIdx.x == 0) { sumOut[c] = sh[0]; ssqOut[c] = sh2[0]; }
}

// ---- BN1 + output linear (WMMA, quad n-tiles) ------------------------------
__global__ __launch_bounds__(32) void bn_outl_kernel(
    const float* __restrict__ X, const float* __restrict__ stats,
    const float* __restrict__ norm_g, const float* __restrict__ norm_b,
    const u16* __restrict__ outlB, const float* __restrict__ outl_b,
    float* __restrict__ Y) {
    __shared__ __align__(16) u16 xnB[16][CC];
    int lane = threadIdx.x;
    int n0 = blockIdx.x * 16;
    int row = lane & 15, half = lane >> 4;
    for (int c = half * 32; c < half * 32 + 32; ++c) {
        float mean = stats[c] * (1.f / NQ);
        float var = stats[64 + c] * (1.f / NQ) - mean * mean;
        float rs = __frsqrt_rn(var + 1e-5f);
        float xv = X[(size_t)(n0 + row) * CC + c];
        xnB[row][c] = f2bf((xv - mean) * rs * norm_g[c] + norm_b[c]);
    }
    __syncthreads();
    {
        float b0 = outl_b[(0 << 4) + (lane & 15)];
        float b1 = outl_b[(1 << 4) + (lane & 15)];
        float b2 = outl_b[(2 << 4) + (lane & 15)];
        float b3 = outl_b[(3 << 4) + (lane & 15)];
        v8f y0, y1, y2, y3;
        gemm_quad(&xnB[0][0], CC, CC, CC, lane,
                  outlB, 0, b0, outlB, 1, b1, outlB, 2, b2, outlB, 3, b3,
                  y0, y1, y2, y3);
        v8f ys[4] = {y0, y1, y2, y3};
#pragma unroll
        for (int nt = 0; nt < 4; ++nt) {
            int col = (nt << 4) + (lane & 15);
            int roff = half * 8;
#pragma unroll
            for (int r = 0; r < 8; ++r) Y[(size_t)(n0 + r + roff) * CC + col] = ys[nt][r];
        }
    }
}

// ---- BN2 + ReLU finalize ----------------------------------------------------
__global__ __launch_bounds__(256) void bn2_relu_kernel(float* __restrict__ Y,
                                                       const float* __restrict__ stats,
                                                       const float* __restrict__ g,
                                                       const float* __restrict__ b) {
    int i = blockIdx.x * 256 + threadIdx.x;
    if (i >= NQ * CC) return;
    int c = i & 63;
    float mean = stats[128 + c] * (1.f / NQ);
    float var = stats[192 + c] * (1.f / NQ) - mean * mean;
    float v = (Y[i] - mean) * __frsqrt_rn(var + 1e-5f) * g[c] + b[c];
    Y[i] = v > 0.f ? v : 0.f;
}

extern "C" void kernel_launch(void* const* d_in, const int* in_sizes, int n_in,
                              void* d_out, int out_size, void* d_ws, size_t ws_size,
                              hipStream_t stream) {
    const float* vf = (const float*)d_in[0];
    const float* vc = (const float*)d_in[1];
    const float* qc = (const float*)d_in[2];
    const int* kidx = (const int*)d_in[3];
    const float* q_w = (const float*)d_in[4];
    const float* q_b = (const float*)d_in[5];
    const float* kpw = (const float*)d_in[6];
    const float* kpb = (const float*)d_in[7];
    const float* in_w = (const float*)d_in[8];
    const float* in_b = (const float*)d_in[9];
    const float* out_w = (const float*)d_in[10];
    const float* out_b = (const float*)d_in[11];
    const float* lin1_w = (const float*)d_in[12];
    const float* lin1_b = (const float*)d_in[13];
    const float* lin2_w = (const float*)d_in[14];
    const float* lin2_b = (const float*)d_in[15];
    const float* norm_g = (const float*)d_in[16];
    const float* norm_b = (const float*)d_in[17];
    const float* outl_w = (const float*)d_in[18];
    const float* outl_b = (const float*)d_in[19];
    const float* obn_g = (const float*)d_in[20];
    const float* obn_b = (const float*)d_in[21];

    // Workspace layout: X (N*64 f32) | stats (256 f32) | bf16 weights (53248 u16)
    float* X = (float*)d_ws;
    float* stats = X + (size_t)NQ * CC;
    u16* wb = (u16*)(stats + 256);
    const u16 *WqB = wb, *WkB = wb + 4096, *WvB = wb + 8192, *WoB = wb + 12288;
    const u16 *outlB = wb + 16384, *l1B = wb + 20480, *l2B = wb + 36864;
    float* Y = (float*)d_out;

    prep_kernel<<<64, 256, 0, stream>>>(in_w, out_w, lin1_w, lin2_w, outl_w, wb);
    attn_fused_kernel<<<NQ / 16, 128, 0, stream>>>(
        vf, vc, qc, kidx, q_w, q_b, kpw, kpb, in_b, out_b, lin1_b, lin2_b,
        WqB, WkB, WvB, WoB, l1B, l2B, X);
    colsum_kernel<<<CC, 256, 0, stream>>>(X, stats, stats + 64);
    bn_outl_kernel<<<NQ / 16, 32, 0, stream>>>(X, stats, norm_g, norm_b, outlB, outl_b, Y);
    colsum_kernel<<<CC, 256, 0, stream>>>(Y, stats + 128, stats + 192);
    bn2_relu_kernel<<<(NQ * CC + 255) / 256, 256, 0, stream>>>(Y, stats, obn_g, obn_b);
}
